// AttentionFusion_15264313770301
// MI455X (gfx1250) — compile-verified, matrix operand order checked by reading, not run on hardware
//
#include <hip/hip_runtime.h>

typedef __bf16 bf16;
typedef __attribute__((ext_vector_type(16))) __bf16 v16bf;
typedef __attribute__((ext_vector_type(8)))  __bf16 v8bf;
typedef __attribute__((ext_vector_type(4)))  __bf16 v4bf;
typedef __attribute__((ext_vector_type(8)))  float  v8f;

#define NEGV (-1.0e10f)

// ---------------------------------------------------------------------------
// Fragment loaders: CDNA5 16-bit A/B layout, wave32.
// lane: idx = lane&15 selects row (A) / col (B); ks = (lane>>4)*8.
// Caller passes pointer to (row_base + ks); elements 0..7 = K ks..ks+7,
// elements 8..15 = K 16+ks..16+ks+7  (two contiguous 16B chunks).
// ---------------------------------------------------------------------------
__device__ __forceinline__ v16bf ldfrag(const bf16* p) {
  v8bf lo = *(const v8bf*)(p);
  v8bf hi = *(const v8bf*)(p + 16);
  v16bf r;
#pragma unroll
  for (int i = 0; i < 8; ++i) { r[i] = lo[i]; r[i + 8] = hi[i]; }
  return r;
}

// fp32 source -> bf16 fragment (on-the-fly convert; avoids a cvt round-trip)
__device__ __forceinline__ v16bf ldfragf(const float* p) {
  float4 a = ((const float4*)p)[0];
  float4 b = ((const float4*)p)[1];
  float4 c = ((const float4*)(p + 16))[0];
  float4 d = ((const float4*)(p + 16))[1];
  v16bf r;
  r[0]  = (bf16)a.x; r[1]  = (bf16)a.y; r[2]  = (bf16)a.z; r[3]  = (bf16)a.w;
  r[4]  = (bf16)b.x; r[5]  = (bf16)b.y; r[6]  = (bf16)b.z; r[7]  = (bf16)b.w;
  r[8]  = (bf16)c.x; r[9]  = (bf16)c.y; r[10] = (bf16)c.z; r[11] = (bf16)c.w;
  r[12] = (bf16)d.x; r[13] = (bf16)d.y; r[14] = (bf16)d.z; r[15] = (bf16)d.w;
  return r;
}

__device__ __forceinline__ v8f wmma_bf16(v16bf a, v16bf b, v8f c) {
  return __builtin_amdgcn_wmma_f32_16x16x32_bf16(
      false, a, false, b, (short)0, c, false, false);
}

// ---------------------------------------------------------------------------
// fp32 -> bf16 elementwise convert (weights only)
// ---------------------------------------------------------------------------
__global__ void __launch_bounds__(256)
cvt_bf16_kernel(const float* __restrict__ in, bf16* __restrict__ out, int n4) {
  int i = blockIdx.x * blockDim.x + threadIdx.x;
  if (i >= n4) return;
  float4 v = ((const float4*)in)[i];
  v4bf o;
  o[0] = (bf16)v.x; o[1] = (bf16)v.y; o[2] = (bf16)v.z; o[3] = (bf16)v.w;
  ((v4bf*)out)[i] = o;
}

// ---------------------------------------------------------------------------
// LayerNorm (torch-clone: Bessel variance, eps added to std), 1 wave/row(256)
// ---------------------------------------------------------------------------
__global__ void __launch_bounds__(256)
ln_bf16_kernel(const float* __restrict__ x, const float* __restrict__ ga,
               const float* __restrict__ gb, bf16* __restrict__ out, int M) {
  int wid  = (blockIdx.x * blockDim.x + threadIdx.x) >> 5;
  int lane = threadIdx.x & 31;
  if (wid >= M) return;
  const float* row = x + (size_t)wid * 256;
  float4 v0 = *(const float4*)(row + lane * 8);
  float4 v1 = *(const float4*)(row + lane * 8 + 4);
  float e[8] = {v0.x, v0.y, v0.z, v0.w, v1.x, v1.y, v1.z, v1.w};
  float s = 0.f;
#pragma unroll
  for (int i = 0; i < 8; ++i) s += e[i];
#pragma unroll
  for (int d = 16; d >= 1; d >>= 1) s += __shfl_xor(s, d, 32);
  float mean = s * (1.0f / 256.0f);
  float ss = 0.f;
#pragma unroll
  for (int i = 0; i < 8; ++i) { float d = e[i] - mean; ss += d * d; }
#pragma unroll
  for (int d = 16; d >= 1; d >>= 1) ss += __shfl_xor(ss, d, 32);
  float var = ss * (1.0f / 255.0f);           // Bessel (n-1)
  float inv = 1.0f / (sqrtf(var) + 1e-6f);    // eps added to std
  v8bf o;
#pragma unroll
  for (int i = 0; i < 8; ++i) {
    int c = lane * 8 + i;
    o[i] = (bf16)(ga[c] * (e[i] - mean) * inv + gb[c]);
  }
  *(v8bf*)(out + (size_t)wid * 256 + lane * 8) = o;
}

// ---------------------------------------------------------------------------
// bf16 WMMA GEMM:  C[M,N] = A[M,K] @ W[N,K]^T  (+bias, relu, scale)
// AF32: A operand is fp32 in memory, converted to bf16 in registers.
// One wave -> 32x64 tile.  Ping-pong double-buffered K loop (step 64,
// K is always a multiple of 64) -- no rotate copies, two fragment sets.
// mode 0: bf16 [M,N] | 1: bf16 [b][h][l][32] | 2: bf16 [b][h][32][l]
// mode 3: fp32 [M,N] + residual
// ---------------------------------------------------------------------------
template <bool AF32>
__global__ void __launch_bounds__(256)
gemm_wmma_kernel(const void* __restrict__ Ap, const bf16* __restrict__ W,
                 const float* __restrict__ bias, const float* __restrict__ resid,
                 void* __restrict__ outp,
                 int M, int N, int K, int L, int Hh,
                 int mode, float scale, int relu) {
  int wid  = (blockIdx.x * blockDim.x + threadIdx.x) >> 5;
  int lane = threadIdx.x & 31;
  int idx  = lane & 15;
  int half = lane >> 4;
  int ks   = half * 8;

  int tiles_n = N >> 6, tiles_m = M >> 5;
  if (wid >= tiles_m * tiles_n) return;
  int tm = wid / tiles_n, tn = wid % tiles_n;
  int m0 = tm * 32, n0 = tn * 64;

  v8f acc[2][4];
#pragma unroll
  for (int a = 0; a < 2; ++a)
#pragma unroll
    for (int b = 0; b < 4; ++b)
#pragma unroll
      for (int j = 0; j < 8; ++j) acc[a][b][j] = 0.f;

  const bf16*  a0p16 = (const bf16*)Ap  + (size_t)(m0 + idx) * K + ks;
  const float* a0p32 = (const float*)Ap + (size_t)(m0 + idx) * K + ks;
  const bf16*  w0p   = W + (size_t)(n0 + idx) * K + ks;

  auto LA0 = [&](int k) -> v16bf {
    if constexpr (AF32) return ldfragf(a0p32 + k);
    else                return ldfrag(a0p16 + k);
  };
  auto LA1 = [&](int k) -> v16bf {
    if constexpr (AF32) return ldfragf(a0p32 + (size_t)16 * K + k);
    else                return ldfrag(a0p16 + (size_t)16 * K + k);
  };
  auto LB = [&](int j, int k) -> v16bf {
    return ldfrag(w0p + (size_t)(16 * j) * K + k);
  };

  // set "c": k, set "n": k+32
  v16bf a0c = LA0(0), a1c = LA1(0), bc[4];
#pragma unroll
  for (int j = 0; j < 4; ++j) bc[j] = LB(j, 0);

  for (int k = 0; k < K; k += 64) {
    // issue loads for k+32 while computing k
    v16bf a0n = LA0(k + 32), a1n = LA1(k + 32), bn[4];
#pragma unroll
    for (int j = 0; j < 4; ++j) bn[j] = LB(j, k + 32);
#pragma unroll
    for (int j = 0; j < 4; ++j) {
      acc[0][j] = wmma_bf16(a0c, bc[j], acc[0][j]);
      acc[1][j] = wmma_bf16(a1c, bc[j], acc[1][j]);
    }
    // issue loads for k+64 while computing k+32
    if (k + 64 < K) {
      a0c = LA0(k + 64); a1c = LA1(k + 64);
#pragma unroll
      for (int j = 0; j < 4; ++j) bc[j] = LB(j, k + 64);
    }
#pragma unroll
    for (int j = 0; j < 4; ++j) {
      acc[0][j] = wmma_bf16(a0n, bn[j], acc[0][j]);
      acc[1][j] = wmma_bf16(a1n, bn[j], acc[1][j]);
    }
  }

#pragma unroll
  for (int mt = 0; mt < 2; ++mt)
#pragma unroll
    for (int nt = 0; nt < 4; ++nt)
#pragma unroll
      for (int j = 0; j < 8; ++j) {
        int m = m0 + mt * 16 + j + 8 * half;   // D layout: VGPR j -> row j+8*half
        int n = n0 + nt * 16 + idx;            // col = idx
        float v = acc[mt][nt][j] + bias[n];
        if (relu) v = fmaxf(v, 0.f);
        v *= scale;
        if (mode == 0) {
          ((bf16*)outp)[(size_t)m * N + n] = (bf16)v;
        } else if (mode == 1) {
          int b = m / L, l = m % L, h = n >> 5, d = n & 31;
          ((bf16*)outp)[(((size_t)(b * Hh + h)) * L + l) * 32 + d] = (bf16)v;
        } else if (mode == 2) {
          int b = m / L, l = m % L, h = n >> 5, d = n & 31;
          ((bf16*)outp)[(((size_t)(b * Hh + h)) * 32 + d) * L + l] = (bf16)v;
        } else {
          float r = resid ? resid[(size_t)m * N + n] : 0.f;
          ((float*)outp)[(size_t)m * N + n] = v + r;
        }
      }
}

// ---------------------------------------------------------------------------
// Flash attention: one wave per (b, h, 16-row q-tile); 32 keys per step
// (dk = 32 == WMMA K).  Single-buffered fragments (no spills) + zero-register
// cache prefetch of the next K/V/mask tiles.  qp pre-scaled by 1/sqrt(dk).
// ---------------------------------------------------------------------------
__global__ void __launch_bounds__(256)
attn_flash_kernel(const bf16* __restrict__ qp, const bf16* __restrict__ kp,
                  const bf16* __restrict__ vpt, const float* __restrict__ mask,
                  bf16* __restrict__ outbf, int Bn, int Hh, int Lq, int Lk) {
  __shared__ bf16  pbuf[8][16 * 32];   // per-wave P (16x32) bf16
  __shared__ float mbuf[8][16 * 32];   // per-wave mask tile (16x32) f32

  int wslot = threadIdx.x >> 5;
  int wid   = (blockIdx.x * blockDim.x + threadIdx.x) >> 5;
  int lane  = threadIdx.x & 31;
  int idx   = lane & 15;
  int half  = lane >> 4;
  int ks    = half * 8;

  int mtiles = Lq >> 4;
  if (wid >= Bn * Hh * mtiles) return;
  int mt = wid % mtiles;
  int bh = wid / mtiles;
  int b  = bh / Hh, h = bh % Hh;
  int m0 = mt << 4;

  bf16*  pb = pbuf[wslot];
  float* mb = mbuf[wslot];

  v16bf qf = ldfrag(qp + ((size_t)bh * Lq + m0 + idx) * 32 + ks);

  v8f o0, o1;
  float mrun[8], lrun[8];
#pragma unroll
  for (int j = 0; j < 8; ++j) { o0[j] = 0.f; o1[j] = 0.f; mrun[j] = -3.0e38f; lrun[j] = 0.f; }

  const bf16* kbase  = kp  + (size_t)bh * Lk * 32;
  const bf16* v0base = vpt + ((size_t)bh * 32 + idx) * Lk + ks;
  const bf16* v1base = v0base + (size_t)16 * Lk;
  const float* mrow  = mask + ((size_t)b * Lq + m0 + (lane >> 1)) * Lk + (lane & 1) * 16;
  float* mdst = mb + (lane >> 1) * 32 + (lane & 1) * 16;

  for (int kt = 0; kt < Lk; kt += 32) {
    // stage 16x32 mask tile: each lane 64 contiguous bytes -> LDS
    const float* mp = mrow + kt;
    float4 f0 = ((const float4*)mp)[0];
    float4 f1 = ((const float4*)mp)[1];
    float4 f2 = ((const float4*)mp)[2];
    float4 f3 = ((const float4*)mp)[3];
    ((float4*)mdst)[0] = f0; ((float4*)mdst)[1] = f1;
    ((float4*)mdst)[2] = f2; ((float4*)mdst)[3] = f3;

    // scores for 32 keys: two 16x16 tiles
    v16bf k0 = ldfrag(kbase + (size_t)(kt + idx) * 32 + ks);
    v16bf k1 = ldfrag(kbase + (size_t)(kt + 16 + idx) * 32 + ks);
    v8f z;
#pragma unroll
    for (int j = 0; j < 8; ++j) z[j] = 0.f;
    v8f s0 = wmma_bf16(qf, k0, z);
    v8f s1 = wmma_bf16(qf, k1, z);

    // zero-register prefetch of next tile (K rows, V rows, mask) into cache
    int kn = kt + 32;
    if (kn < Lk) {
      __builtin_prefetch(kbase + (size_t)(kn + idx) * 32, 0, 3);
      __builtin_prefetch(kbase + (size_t)(kn + 16 + idx) * 32, 0, 3);
      __builtin_prefetch(v0base + kn, 0, 3);
      __builtin_prefetch(v1base + kn, 0, 3);
      __builtin_prefetch(mrow + kn, 0, 3);
    }

    asm volatile("s_wait_dscnt 0x0" ::: "memory");  // mask tile visible

    float p0[8], p1[8], tmx[8];
#pragma unroll
    for (int j = 0; j < 8; ++j) {
      float a = s0[j], c = s1[j];
      if (mb[(j + 8 * half) * 32 + idx]      == 0.f) a = NEGV;
      if (mb[(j + 8 * half) * 32 + 16 + idx] == 0.f) c = NEGV;
      p0[j] = a; p1[j] = c;
      tmx[j] = fmaxf(a, c);
    }
    for (int s = 1; s < 16; s <<= 1) {
#pragma unroll
      for (int j = 0; j < 8; ++j) tmx[j] = fmaxf(tmx[j], __shfl_xor(tmx[j], s, 32));
    }
    float rs[8], alpha[8];
#pragma unroll
    for (int j = 0; j < 8; ++j) {
      float mn = fmaxf(mrun[j], tmx[j]);
      alpha[j] = __expf(mrun[j] - mn);
      mrun[j]  = mn;
      p0[j] = __expf(p0[j] - mn);
      p1[j] = __expf(p1[j] - mn);
      rs[j] = p0[j] + p1[j];
    }
    for (int s = 1; s < 16; s <<= 1) {
#pragma unroll
      for (int j = 0; j < 8; ++j) rs[j] += __shfl_xor(rs[j], s, 32);
    }
#pragma unroll
    for (int j = 0; j < 8; ++j) {
      lrun[j] = lrun[j] * alpha[j] + rs[j];
      o0[j] *= alpha[j];
      o1[j] *= alpha[j];
      pb[(j + 8 * half) * 32 + idx]      = (bf16)p0[j];
      pb[(j + 8 * half) * 32 + 16 + idx] = (bf16)p1[j];
    }
    asm volatile("s_wait_dscnt 0x0" ::: "memory");  // P tile visible

    v16bf pf  = ldfrag(pb + idx * 32 + ks);  // P as A-fragment (16x32)
    v16bf vb0 = ldfrag(v0base + kt);         // V^T rows d=idx
    v16bf vb1 = ldfrag(v1base + kt);         // V^T rows d=16+idx
    o0 = wmma_bf16(pf, vb0, o0);
    o1 = wmma_bf16(pf, vb1, o1);
  }

#pragma unroll
  for (int j = 0; j < 8; ++j) {
    float inv = 1.0f / lrun[j];
    int m = m0 + j + 8 * half;
    size_t rowo = ((size_t)b * Lq + m) * 256 + (size_t)h * 32;
    outbf[rowo + idx]      = (bf16)(o0[j] * inv);
    outbf[rowo + 16 + idx] = (bf16)(o1[j] * inv);
  }
}

// ---------------------------------------------------------------------------
// Host-side orchestration
// ---------------------------------------------------------------------------
static inline int gemm_grid(int M, int N) {
  int waves = (M >> 5) * (N >> 6);
  return (waves + 7) / 8;   // 8 waves (256 threads) per block
}

extern "C" void kernel_launch(void* const* d_in, const int* in_sizes, int n_in,
                              void* d_out, int out_size, void* d_ws, size_t ws_size,
                              hipStream_t stream) {
  const float* q    = (const float*)d_in[0];
  const float* k    = (const float*)d_in[1];
  const float* v    = (const float*)d_in[2];
  const float* mask = (const float*)d_in[3];
  const float* wq   = (const float*)d_in[4];  const float* bq = (const float*)d_in[5];
  const float* wk   = (const float*)d_in[6];  const float* bk = (const float*)d_in[7];
  const float* wv   = (const float*)d_in[8];  const float* bv = (const float*)d_in[9];
  const float* wo   = (const float*)d_in[10]; const float* bo = (const float*)d_in[11];
  const float* w1   = (const float*)d_in[12]; const float* b1 = (const float*)d_in[13];
  const float* w2   = (const float*)d_in[14]; const float* b2 = (const float*)d_in[15];
  const float* l1a  = (const float*)d_in[16]; const float* l1b = (const float*)d_in[17];
  const float* l2a  = (const float*)d_in[18]; const float* l2b = (const float*)d_in[19];

  const int Bn = 16, H = 8, Lq = 1024, Lk = 3072, D = 256, F = 1024;
  const int Mq = Bn * Lq;   // 16384
  const int Mk = Bn * Lk;   // 49152

  char* ws = (char*)d_ws;
  size_t off = 0;
  auto alloc = [&](size_t bytes) -> char* {
    char* p = ws + off;
    off += (bytes + 255) & ~(size_t)255;
    return p;
  };

  bf16* wq_b  = (bf16*)alloc((size_t)D * D * 2);
  bf16* wk_b  = (bf16*)alloc((size_t)D * D * 2);
  bf16* wv_b  = (bf16*)alloc((size_t)D * D * 2);
  bf16* wo_b  = (bf16*)alloc((size_t)D * D * 2);
  bf16* w1_b  = (bf16*)alloc((size_t)F * D * 2);
  bf16* w2_b  = (bf16*)alloc((size_t)D * F * 2);
  bf16* lnq_b = (bf16*)alloc((size_t)Mq * D * 2);   //  8 MB -> attn_b reuse
  bf16* qp_b  = (bf16*)alloc((size_t)Mq * D * 2);   //  8 MB }
  bf16* kp_b  = (bf16*)alloc((size_t)Mk * D * 2);   // 24 MB } -> h_b (32MB) reuse
  bf16* vpt_b = (bf16*)alloc((size_t)Mk * D * 2);   // 24 MB -> x1 + ln2_b reuse

  bf16*  attn_b = lnq_b;                              // dead after Q projection
  float* x1     = (float*)vpt_b;                      // dead after attention (16MB)
  bf16*  ln2_b  = (bf16*)((char*)vpt_b + (size_t)Mq * D * 4);  // +16MB, 8MB
  bf16*  h_b    = qp_b;                               // qp+kp dead after attention

  const int THR = 256;
  const float inv_sqrt_dk = 0.17677669529663689f;     // 1/sqrt(32)

  // 1) weights fp32 -> bf16
  {
    int n4 = D * D / 4;
    cvt_bf16_kernel<<<(n4 + THR - 1) / THR, THR, 0, stream>>>(wq, wq_b, n4);
    cvt_bf16_kernel<<<(n4 + THR - 1) / THR, THR, 0, stream>>>(wk, wk_b, n4);
    cvt_bf16_kernel<<<(n4 + THR - 1) / THR, THR, 0, stream>>>(wv, wv_b, n4);
    cvt_bf16_kernel<<<(n4 + THR - 1) / THR, THR, 0, stream>>>(wo, wo_b, n4);
    n4 = F * D / 4;
    cvt_bf16_kernel<<<(n4 + THR - 1) / THR, THR, 0, stream>>>(w1, w1_b, n4);
    cvt_bf16_kernel<<<(n4 + THR - 1) / THR, THR, 0, stream>>>(w2, w2_b, n4);
  }

  // 2) LN1(q) -> bf16
  ln_bf16_kernel<<<(Mq * 32 + THR - 1) / THR, THR, 0, stream>>>(q, l1a, l1b, lnq_b, Mq);

  // 3) Q projection (pre-scaled by 1/sqrt(dk)), per-head layout
  gemm_wmma_kernel<false><<<gemm_grid(Mq, D), THR, 0, stream>>>(
      lnq_b, wq_b, bq, nullptr, qp_b, Mq, D, D, Lq, H, 1, inv_sqrt_dk, 0);
  // 4) K projection: fp32 A read directly, per-head layout
  gemm_wmma_kernel<true><<<gemm_grid(Mk, D), THR, 0, stream>>>(
      k, wk_b, bk, nullptr, kp_b, Mk, D, D, Lk, H, 1, 1.0f, 0);
  // 5) V projection: fp32 A read directly, per-head transposed layout
  gemm_wmma_kernel<true><<<gemm_grid(Mk, D), THR, 0, stream>>>(
      v, wv_b, bv, nullptr, vpt_b, Mk, D, D, Lk, H, 2, 1.0f, 0);

  // 6) Flash attention -> concat-head bf16 [Mq, 256]
  {
    int waves = Bn * H * (Lq >> 4);          // 8192
    attn_flash_kernel<<<(waves + 7) / 8, THR, 0, stream>>>(
        qp_b, kp_b, vpt_b, mask, attn_b, Bn, H, Lq, Lk);
  }

  // 7) Output projection + residual(q) -> fp32 x1
  gemm_wmma_kernel<false><<<gemm_grid(Mq, D), THR, 0, stream>>>(
      attn_b, wo_b, bo, q, x1, Mq, D, D, Lq, H, 3, 1.0f, 0);

  // 8) LN2(x1) -> bf16
  ln_bf16_kernel<<<(Mq * 32 + THR - 1) / THR, THR, 0, stream>>>(x1, l2a, l2b, ln2_b, Mq);

  // 9) FFN up + relu -> bf16 h
  gemm_wmma_kernel<false><<<gemm_grid(Mq, F), THR, 0, stream>>>(
      ln2_b, w1_b, b1, nullptr, h_b, Mq, F, D, Lq, H, 0, 1.0f, 1);

  // 10) FFN down + residual(x1) -> fp32 d_out
  gemm_wmma_kernel<false><<<gemm_grid(Mq, D), THR, 0, stream>>>(
      h_b, w2_b, b2, x1, d_out, Mq, D, F, Lq, H, 3, 1.0f, 0);

  (void)in_sizes; (void)n_in; (void)out_size; (void)ws_size;
}
